// AttentionLstmDecoderV1_75230647156891
// MI455X (gfx1250) — matrix operand order repaired and checked by reading, not compiled, not run on hardware
//
#include <hip/hip_runtime.h>
#include <hip/hip_bf16.h>
#include <math.h>

// Problem constants (match the reference)
#define B_      16
#define T_      1000
#define N_      100
#define ENC_    512
#define VOCAB_  10025
#define EMB_    640
#define HID_    1024
#define ATT_    1024
#define PROJ_   1024
#define XDIM_   (EMB_ + ENC_)          // 1152
#define RODIM_  (HID_ + EMB_ + ENC_)   // 2176

typedef __attribute__((ext_vector_type(16))) __bf16 v16bf;
typedef __attribute__((ext_vector_type(8)))  float  v8f;
typedef __attribute__((address_space(3)))    float  lds_f32;

// ---------------------------------------------------------------------------
// fp32 -> bf16 weight conversion
// ---------------------------------------------------------------------------
__global__ void cvt_bf16_kernel(const float* __restrict__ src,
                                __bf16* __restrict__ dst, int n) {
  int i = blockIdx.x * blockDim.x + threadIdx.x;
  if (i < n) dst[i] = (__bf16)src[i];
}

__global__ void zero_kernel(float* __restrict__ p, int n) {
  int i = blockIdx.x * blockDim.x + threadIdx.x;
  if (i < n) p[i] = 0.f;
}

__global__ void bias_sum_kernel(const float* __restrict__ a,
                                const float* __restrict__ b,
                                float* __restrict__ o, int n) {
  int i = blockIdx.x * blockDim.x + threadIdx.x;
  if (i < n) o[i] = a[i] + b[i];
}

// ---------------------------------------------------------------------------
// Generic WMMA GEMM:  C[M,N] (+)= A[M,K]_f32  x  W[N,K]_bf16^T  (+ bias[N])
//
// - one wave computes one 16x16 tile; 8 waves per 256-thread block tile N
// - the shared 16x64 fp32 A tile is staged into LDS via the CDNA5 async
//   DMA path (GLOBAL_LOAD_ASYNC_TO_LDS_B128, ASYNCcnt), double-buffered so
//   the next k-tile's DMA overlaps the current tile's WMMAs
// - A converted fp32->bf16 in-register at fragment build (v_cvt_pk_bf16_f32)
// - fragment layouts: A and B share contiguous K order k = 16*half + j (16-bit A 16x32, B 32x16)
// ---------------------------------------------------------------------------
__global__ void __launch_bounds__(256)
gemm_bf16_kernel(const float* __restrict__ A, int lda,
                 const __bf16* __restrict__ W, int ldw,
                 float* __restrict__ C, int ldc,
                 const float* __restrict__ bias,
                 int M, int N, int K, int accum)
{
  __shared__ float atile[2][16][64];     // double-buffered 16x64 fp32 A tile

  const int tid  = threadIdx.x;
  const int lane = tid & 31;
  const int wave = tid >> 5;
  const int n0 = (blockIdx.x * 8 + wave) * 16;
  const int m0 = blockIdx.y * 16;
  const bool active = (n0 < N);          // no early return: LDS barriers below

  const int half = lane >> 4;            // K-half selector (A/B), M-half (C/D)
  const int l15  = lane & 15;            // A row within tile; B/D column

  const int ncol = n0 + l15;
  const int ncl  = (ncol < N) ? ncol : (N - 1);   // clamped for safe loads

  v8f acc = {0.f, 0.f, 0.f, 0.f, 0.f, 0.f, 0.f, 0.f};
  if (accum) {
#pragma unroll
    for (int r = 0; r < 8; ++r) {
      int mm = m0 + r + 8 * half;
      acc[r] = (active && ncol < N && mm < M) ? C[(size_t)mm * ldc + ncol] : 0.f;
    }
  }

  const __bf16* wrow = W + (size_t)ncl * ldw + half * 16;

  // Cooperative A staging: 256 threads x 16B cover the 16x64 fp32 tile.
  const int srow   = tid >> 4;           // 0..15   (tile row)
  const int schunk = (tid & 15) << 2;    // 0..60   (float offset, 16B chunks)
  const float* gsrc = A + (size_t)(m0 + srow) * lda + schunk;

  // Prologue: async-DMA k-tile 0 into buffer 0.
  {
    unsigned loff = (unsigned)(size_t)(lds_f32*)&atile[0][srow][schunk];
    asm volatile("global_load_async_to_lds_b128 %0, %1, off"
                 :: "v"(loff), "v"(gsrc) : "memory");
  }

  const int nk = K >> 6;                 // K is always a multiple of 64 here
  for (int kt = 0; kt < nk; ++kt) {
    asm volatile("s_wait_asynccnt 0" ::: "memory");  // my slice of tile landed
    __syncthreads();                                 // whole tile landed

    if (kt + 1 < nk) {                   // overlap next DMA with compute
      unsigned loff =
          (unsigned)(size_t)(lds_f32*)&atile[(kt + 1) & 1][srow][schunk];
      const float* g = gsrc + (size_t)(kt + 1) * 64;
      asm volatile("global_load_async_to_lds_b128 %0, %1, off"
                   :: "v"(loff), "v"(g) : "memory");
    }

    const float* arow_l = &atile[kt & 1][l15][0];
    __builtin_prefetch(wrow + (size_t)kt * 64 + 128, 0, 1);  // global_prefetch_b8

#pragma unroll
    for (int ks = 0; ks < 2; ++ks) {
      // A fragment: lanes 0-15 rows M=0..15 hold K 0..15;
      //             lanes 16-31 hold K 16..31
      v16bf afrag;
#pragma unroll
      for (int j = 0; j < 16; ++j) {
        int kk = ks * 32 + half * 16 + j;
        afrag[j] = (__bf16)arow_l[kk];
      }
      // B fragment: column n = lane&15; lanes 0-15 hold K=0..15,
      // lanes 16-31 K=16..31 — contiguous 32B load from [N,K] bf16 weights.
      v16bf bfrag = *(const v16bf*)(wrow + (size_t)kt * 64 + ks * 32);

      acc = __builtin_amdgcn_wmma_f32_16x16x32_bf16(
          /*neg_a=*/false, afrag, /*neg_b=*/false, bfrag,
          /*c_mod=*/(short)0, acc, /*reuse_a=*/false, /*reuse_b=*/false);
    }
    __syncthreads();                     // all waves done reading this buffer
  }

  if (active && ncol < N) {
    float bi = bias ? bias[ncol] : 0.f;
#pragma unroll
    for (int r = 0; r < 8; ++r) {
      int mm = m0 + r + 8 * half;
      if (mm < M) C[(size_t)mm * ldc + ncol] = acc[r] + bi;
    }
  }
}

// ---------------------------------------------------------------------------
// Shifted target embeddings: emb[b,n,:] = n==0 ? 0 : table[labels[b,n-1],:]
// ---------------------------------------------------------------------------
__global__ void embed_kernel(const float* __restrict__ table,
                             const int* __restrict__ labels,
                             float* __restrict__ emb) {
  size_t i = (size_t)blockIdx.x * blockDim.x + threadIdx.x;
  if (i >= (size_t)B_ * N_ * EMB_) return;
  int j = (int)(i % EMB_);
  int n = (int)((i / EMB_) % N_);
  int b = (int)(i / ((size_t)EMB_ * N_));
  emb[i] = (n == 0) ? 0.f
                    : table[(size_t)labels[b * N_ + n - 1] * EMB_ + j];
}

// ---------------------------------------------------------------------------
// inv_fertility: sigmoid(enc_out[b,t,:] . W_fert)  — one wave per (b,t) row
// ---------------------------------------------------------------------------
__global__ void __launch_bounds__(256)
inv_fert_kernel(const float* __restrict__ enc_out,
                const float* __restrict__ W_fert,
                float* __restrict__ inv_fert) {
  int row  = blockIdx.x * 8 + (threadIdx.x >> 5);
  int lane = threadIdx.x & 31;
  if (row >= B_ * T_) return;
  const float* e = enc_out + (size_t)row * ENC_;
  float p = 0.f;
  for (int j = lane; j < ENC_; j += 32) p += e[j] * W_fert[j];
#pragma unroll
  for (int off = 16; off > 0; off >>= 1) p += __shfl_xor(p, off, 32);
  if (lane == 0) inv_fert[row] = 1.f / (1.f + __expf(-p));
}

// ---------------------------------------------------------------------------
// Per-step LSTM input x = [emb[:,step,:], att_ctx]
// ---------------------------------------------------------------------------
__global__ void build_x_kernel(const float* __restrict__ emb,
                               const float* __restrict__ att_ctx,
                               float* __restrict__ x, int step) {
  int i = blockIdx.x * blockDim.x + threadIdx.x;
  if (i >= B_ * XDIM_) return;
  int b = i / XDIM_, j = i % XDIM_;
  x[i] = (j < EMB_) ? emb[((size_t)b * N_ + step) * EMB_ + j]
                    : att_ctx[b * ENC_ + (j - EMB_)];
}

// ---------------------------------------------------------------------------
// LSTM pointwise: gates [B,4H] -> h,c; also stash h into s_seq[b,step,:]
// ---------------------------------------------------------------------------
__global__ void lstm_pointwise_kernel(const float* __restrict__ gates,
                                      float* __restrict__ h,
                                      float* __restrict__ c,
                                      float* __restrict__ s_seq, int step) {
  int i = blockIdx.x * blockDim.x + threadIdx.x;
  if (i >= B_ * HID_) return;
  int b = i / HID_, j = i % HID_;
  const float* g = gates + (size_t)b * 4 * HID_;
  float gi = g[j], gf = g[HID_ + j], gg = g[2 * HID_ + j], go = g[3 * HID_ + j];
  float si = 1.f / (1.f + __expf(-gi));
  float sf = 1.f / (1.f + __expf(-gf));
  float so = 1.f / (1.f + __expf(-go));
  float cn = sf * c[i] + si * tanhf(gg);
  float hn = so * tanhf(cn);
  c[i] = cn;
  h[i] = hn;
  s_seq[((size_t)b * N_ + step) * HID_ + j] = hn;
}

// ---------------------------------------------------------------------------
// Attention energies: e[b,t] = v . tanh(enc_ctx[b,t,:] + s_t[b,:] + accum*W_fb)
// grid (B, 8): each block handles a 125-step t-chunk; block reduction over ATT
// ---------------------------------------------------------------------------
__global__ void __launch_bounds__(256)
attn_energy_kernel(const float* __restrict__ enc_ctx,
                   const float* __restrict__ s_t,
                   const float* __restrict__ accum,
                   const float* __restrict__ W_fb,
                   const float* __restrict__ v_att,
                   const int* __restrict__ seq_len,
                   float* __restrict__ energies) {
  __shared__ float red[256];
  const int b = blockIdx.x;
  const int tid = threadIdx.x;
  const int len = seq_len[b];
  const float* ss = s_t + b * ATT_;
  for (int t = blockIdx.y * 125; t < (blockIdx.y + 1) * 125; ++t) {
    const float ac = accum[b * T_ + t];
    const float* ec = enc_ctx + ((size_t)b * T_ + t) * ATT_;
    float p = 0.f;
    for (int j = tid; j < ATT_; j += 256)
      p += v_att[j] * tanhf(ec[j] + ss[j] + ac * W_fb[j]);
    red[tid] = p;
    __syncthreads();
    for (int s = 128; s > 0; s >>= 1) {
      if (tid < s) red[tid] += red[tid + s];
      __syncthreads();
    }
    if (tid == 0) energies[b * T_ + t] = (t < len) ? red[0] : -1e30f;
    __syncthreads();
  }
}

// ---------------------------------------------------------------------------
// Softmax over T (masked via -1e30), accum update, context vector.
// One block per batch element; enc_out rows stream at L2 bandwidth.
// ---------------------------------------------------------------------------
__global__ void __launch_bounds__(512)
attn_softmax_ctx_kernel(const float* __restrict__ enc_out,
                        const float* __restrict__ inv_fert,
                        float* __restrict__ energies,   // in: e; out: exp(e-max)
                        float* __restrict__ accum,
                        float* __restrict__ att_ctx,
                        float* __restrict__ ctx_seq, int step) {
  __shared__ float red[512];
  __shared__ float s_max, s_rinv;
  const int b = blockIdx.x;
  const int tid = threadIdx.x;

  float m = -1e30f;
  for (int t = tid; t < T_; t += 512) m = fmaxf(m, energies[b * T_ + t]);
  red[tid] = m; __syncthreads();
  for (int s = 256; s > 0; s >>= 1) {
    if (tid < s) red[tid] = fmaxf(red[tid], red[tid + s]);
    __syncthreads();
  }
  if (tid == 0) s_max = red[0];
  __syncthreads();

  float p = 0.f;
  for (int t = tid; t < T_; t += 512) {
    float w = __expf(energies[b * T_ + t] - s_max);
    energies[b * T_ + t] = w;
    p += w;
  }
  red[tid] = p; __syncthreads();
  for (int s = 256; s > 0; s >>= 1) {
    if (tid < s) red[tid] += red[tid + s];
    __syncthreads();
  }
  if (tid == 0) s_rinv = 1.f / red[0];
  __syncthreads();
  const float rinv = s_rinv;

  for (int t = tid; t < T_; t += 512)
    accum[b * T_ + t] += energies[b * T_ + t] * rinv * inv_fert[b * T_ + t] * 0.5f;

  if (tid < ENC_) {
    float cj = 0.f;
    const float* eb = enc_out + (size_t)b * T_ * ENC_ + tid;
    const float* wb = energies + b * T_;
    for (int t = 0; t < T_; ++t) cj += wb[t] * eb[(size_t)t * ENC_];
    cj *= rinv;
    att_ctx[b * ENC_ + tid] = cj;
    ctx_seq[((size_t)b * N_ + step) * ENC_ + tid] = cj;
  }
}

// ---------------------------------------------------------------------------
// Readout input concat, maxout, and tail copy
// ---------------------------------------------------------------------------
__global__ void concat_ro_kernel(const float* __restrict__ s_seq,
                                 const float* __restrict__ emb,
                                 const float* __restrict__ ctx_seq,
                                 float* __restrict__ ro_in) {
  size_t i = (size_t)blockIdx.x * blockDim.x + threadIdx.x;
  if (i >= (size_t)B_ * N_ * RODIM_) return;
  size_t row = i / RODIM_;
  int j = (int)(i % RODIM_);
  float v;
  if (j < HID_)             v = s_seq[row * HID_ + j];
  else if (j < HID_ + EMB_) v = emb[row * EMB_ + (j - HID_)];
  else                      v = ctx_seq[row * ENC_ + (j - HID_ - EMB_)];
  ro_in[i] = v;
}

__global__ void maxout_kernel(const float* __restrict__ ro_full,
                              float* __restrict__ readout) {
  size_t i = (size_t)blockIdx.x * blockDim.x + threadIdx.x;
  if (i >= (size_t)B_ * N_ * (PROJ_ / 2)) return;
  size_t row = i / (PROJ_ / 2);
  int j = (int)(i % (PROJ_ / 2));
  const float* r = ro_full + row * PROJ_;
  readout[i] = fmaxf(r[2 * j], r[2 * j + 1]);
}

__global__ void tail_kernel(const float* __restrict__ h,
                            const float* __restrict__ c,
                            const float* __restrict__ att_ctx,
                            const float* __restrict__ accum,
                            float* __restrict__ out_tail) {
  int i = blockIdx.x * blockDim.x + threadIdx.x;
  const int n1 = B_ * HID_;
  const int n2 = 2 * n1;
  const int n3 = n2 + B_ * ENC_;
  const int n4 = n3 + B_ * T_;
  if (i < n1)      out_tail[i] = h[i];
  else if (i < n2) out_tail[i] = c[i - n1];
  else if (i < n3) out_tail[i] = att_ctx[i - n2];
  else if (i < n4) out_tail[i] = accum[i - n3];
}

// ---------------------------------------------------------------------------
// Host launcher
// ---------------------------------------------------------------------------
extern "C" void kernel_launch(void* const* d_in, const int* in_sizes, int n_in,
                              void* d_out, int out_size, void* d_ws, size_t ws_size,
                              hipStream_t stream) {
  (void)in_sizes; (void)n_in; (void)out_size; (void)ws_size;

  const float* enc_out   = (const float*)d_in[0];
  const int*   labels    = (const int*)  d_in[1];
  const int*   seq_len   = (const int*)  d_in[2];
  const float* table     = (const float*)d_in[3];
  const float* W_ih      = (const float*)d_in[4];
  const float* W_hh      = (const float*)d_in[5];
  const float* b_ih      = (const float*)d_in[6];
  const float* b_hh      = (const float*)d_in[7];
  const float* W_s       = (const float*)d_in[8];
  const float* W_enc     = (const float*)d_in[9];
  const float* b_enc     = (const float*)d_in[10];
  const float* v_att     = (const float*)d_in[11];
  const float* W_fert    = (const float*)d_in[12];
  const float* W_fb      = (const float*)d_in[13];
  const float* W_ro      = (const float*)d_in[14];
  const float* b_ro      = (const float*)d_in[15];
  const float* W_out     = (const float*)d_in[16];
  const float* b_out     = (const float*)d_in[17];
  float* out = (float*)d_out;

  char* ws = (char*)d_ws;
  size_t off = 0;
  auto alloc = [&](size_t bytes) -> size_t {
    size_t o = off;
    off += (bytes + 255) & ~(size_t)255;
    return o;
  };

  float*  enc_ctx  = (float*)(ws + alloc((size_t)B_ * T_ * ATT_ * 4));
  float*  inv_fert = (float*)(ws + alloc((size_t)B_ * T_ * 4));
  float*  emb      = (float*)(ws + alloc((size_t)B_ * N_ * EMB_ * 4));
  float*  bsum     = (float*)(ws + alloc((size_t)4 * HID_ * 4));
  __bf16* Wih_bf   = (__bf16*)(ws + alloc((size_t)4 * HID_ * XDIM_ * 2));
  __bf16* Whh_bf   = (__bf16*)(ws + alloc((size_t)4 * HID_ * HID_ * 2));
  __bf16* Ws_bf    = (__bf16*)(ws + alloc((size_t)ATT_ * HID_ * 2));
  __bf16* Wenc_bf  = (__bf16*)(ws + alloc((size_t)ATT_ * ENC_ * 2));
  __bf16* Wro_bf   = (__bf16*)(ws + alloc((size_t)PROJ_ * RODIM_ * 2));
  __bf16* Wout_bf  = (__bf16*)(ws + alloc((size_t)VOCAB_ * (PROJ_ / 2) * 2));
  float*  h        = (float*)(ws + alloc((size_t)B_ * HID_ * 4));
  float*  c        = (float*)(ws + alloc((size_t)B_ * HID_ * 4));
  float*  att_ctx  = (float*)(ws + alloc((size_t)B_ * ENC_ * 4));
  float*  accum    = (float*)(ws + alloc((size_t)B_ * T_ * 4));
  float*  x        = (float*)(ws + alloc((size_t)B_ * XDIM_ * 4));
  float*  gates    = (float*)(ws + alloc((size_t)B_ * 4 * HID_ * 4));
  float*  s_t      = (float*)(ws + alloc((size_t)B_ * ATT_ * 4));
  float*  energies = (float*)(ws + alloc((size_t)B_ * T_ * 4));
  float*  s_seq    = (float*)(ws + alloc((size_t)B_ * N_ * HID_ * 4));
  float*  ctx_seq  = (float*)(ws + alloc((size_t)B_ * N_ * ENC_ * 4));
  float*  ro_in    = (float*)(ws + alloc((size_t)B_ * N_ * RODIM_ * 4));
  float*  ro_full  = (float*)(ws + alloc((size_t)B_ * N_ * PROJ_ * 4));
  float*  readout  = (float*)(ws + alloc((size_t)B_ * N_ * (PROJ_ / 2) * 4));

  auto cgrid = [](size_t n) { return dim3((unsigned)((n + 255) / 256)); };

  // --- weight conversions to bf16 ---
  {
    struct { const float* s; __bf16* d; size_t n; } cv[] = {
      {W_ih,  Wih_bf,  (size_t)4 * HID_ * XDIM_},
      {W_hh,  Whh_bf,  (size_t)4 * HID_ * HID_},
      {W_s,   Ws_bf,   (size_t)ATT_ * HID_},
      {W_enc, Wenc_bf, (size_t)ATT_ * ENC_},
      {W_ro,  Wro_bf,  (size_t)PROJ_ * RODIM_},
      {W_out, Wout_bf, (size_t)VOCAB_ * (PROJ_ / 2)},
    };
    for (auto& e : cv)
      cvt_bf16_kernel<<<cgrid(e.n), 256, 0, stream>>>(e.s, e.d, (int)e.n);
  }
  bias_sum_kernel<<<cgrid(4 * HID_), 256, 0, stream>>>(b_ih, b_hh, bsum, 4 * HID_);

  // --- state init (ws is poisoned before timing) ---
  zero_kernel<<<cgrid(B_ * HID_), 256, 0, stream>>>(h, B_ * HID_);
  zero_kernel<<<cgrid(B_ * HID_), 256, 0, stream>>>(c, B_ * HID_);
  zero_kernel<<<cgrid(B_ * ENC_), 256, 0, stream>>>(att_ctx, B_ * ENC_);
  zero_kernel<<<cgrid(B_ * T_), 256, 0, stream>>>(accum, B_ * T_);

  // --- shifted embeddings, inv_fertility, enc_ctx GEMM ---
  embed_kernel<<<cgrid((size_t)B_ * N_ * EMB_), 256, 0, stream>>>(table, labels, emb);
  inv_fert_kernel<<<dim3((B_ * T_ + 7) / 8), 256, 0, stream>>>(enc_out, W_fert, inv_fert);
  // enc_ctx[B*T, ATT] = enc_out[B*T, ENC] @ W_enc^T + b_enc
  gemm_bf16_kernel<<<dim3((ATT_ / 16 + 7) / 8, B_ * T_ / 16), 256, 0, stream>>>(
      enc_out, ENC_, Wenc_bf, ENC_, enc_ctx, ATT_, b_enc,
      B_ * T_, ATT_, ENC_, 0);

  // --- decoder scan: N steps, stream-ordered ---
  for (int step = 0; step < N_; ++step) {
    build_x_kernel<<<cgrid(B_ * XDIM_), 256, 0, stream>>>(emb, att_ctx, x, step);
    // gates = x @ W_ih^T + (b_ih + b_hh)
    gemm_bf16_kernel<<<dim3((4 * HID_ / 16 + 7) / 8, 1), 256, 0, stream>>>(
        x, XDIM_, Wih_bf, XDIM_, gates, 4 * HID_, bsum, B_, 4 * HID_, XDIM_, 0);
    // gates += h @ W_hh^T
    gemm_bf16_kernel<<<dim3((4 * HID_ / 16 + 7) / 8, 1), 256, 0, stream>>>(
        h, HID_, Whh_bf, HID_, gates, 4 * HID_, nullptr, B_, 4 * HID_, HID_, 1);
    lstm_pointwise_kernel<<<cgrid(B_ * HID_), 256, 0, stream>>>(gates, h, c, s_seq, step);
    // s_t = h @ W_s^T
    gemm_bf16_kernel<<<dim3((ATT_ / 16 + 7) / 8, 1), 256, 0, stream>>>(
        h, HID_, Ws_bf, HID_, s_t, ATT_, nullptr, B_, ATT_, HID_, 0);
    attn_energy_kernel<<<dim3(B_, 8), 256, 0, stream>>>(
        enc_ctx, s_t, accum, W_fb, v_att, seq_len, energies);
    attn_softmax_ctx_kernel<<<dim3(B_), 512, 0, stream>>>(
        enc_out, inv_fert, energies, accum, att_ctx, ctx_seq, step);
  }

  // --- readout: concat -> GEMM -> maxout -> vocab GEMM (into d_out) ---
  concat_ro_kernel<<<cgrid((size_t)B_ * N_ * RODIM_), 256, 0, stream>>>(
      s_seq, emb, ctx_seq, ro_in);
  gemm_bf16_kernel<<<dim3((PROJ_ / 16 + 7) / 8, B_ * N_ / 16), 256, 0, stream>>>(
      ro_in, RODIM_, Wro_bf, RODIM_, ro_full, PROJ_, b_ro,
      B_ * N_, PROJ_, RODIM_, 0);
  maxout_kernel<<<cgrid((size_t)B_ * N_ * (PROJ_ / 2)), 256, 0, stream>>>(ro_full, readout);
  // logits[B*N, VOCAB] straight into d_out
  gemm_bf16_kernel<<<dim3(((VOCAB_ + 15) / 16 + 7) / 8, B_ * N_ / 16), 256, 0, stream>>>(
      readout, PROJ_ / 2, Wout_bf, PROJ_ / 2, out, VOCAB_, b_out,
      B_ * N_, VOCAB_, PROJ_ / 2, 0);

  // --- tail outputs: h, c, att_ctx, accum after logits ---
  float* out_tail = out + (size_t)B_ * N_ * VOCAB_;
  int tail_n = 2 * B_ * HID_ + B_ * ENC_ + B_ * T_;
  tail_kernel<<<cgrid(tail_n), 256, 0, stream>>>(h, c, att_ctx, accum, out_tail);
}
